// EfficientSelfAttention_47991964565899
// MI455X (gfx1250) — compile-verified, matrix operand order checked by reading, not run on hardware
//
#include <hip/hip_runtime.h>
#include <hip/hip_bf16.h>

// ---------------------------------------------------------------------------
// EfficientSelfAttention (PVT spatial-reduction attention) for MI455X/gfx1250
// B=8, N=4096, C=512, NH=8, hd=64, R=4, H=W=64, Nr=256
// All heavy math through v_wmma_f32_16x16x32_bf16 (wave32).
// GEMM B-tiles staged to LDS via gfx1250 async global->LDS copies.
// ---------------------------------------------------------------------------

typedef __bf16 bf16_t;
typedef __attribute__((ext_vector_type(16))) __bf16 v16bf;
typedef __attribute__((ext_vector_type(8)))  __bf16 v8bf;
typedef __attribute__((ext_vector_type(8)))  float  v8f;

// GCC-style vector matching the async-to-LDS builtin's parameter type
typedef int v4i_t __attribute__((vector_size(4 * sizeof(int))));
typedef __attribute__((address_space(1))) v4i_t* gptr_v4i;
typedef __attribute__((address_space(3))) v4i_t* lptr_v4i;

#define BB 8
#define NN 4096
#define CC 512
#define NHH 8
#define HD 64
#define NR 256
#define KHW 16   // R*R

#if __has_builtin(__builtin_amdgcn_global_load_async_to_lds_b128)
#define HAVE_ASYNC_LDS 1
#else
#define HAVE_ASYNC_LDS 0
#endif

static __device__ __forceinline__ v16bf load_frag(const bf16_t* p) {
  // ISA 16-bit A/B fragment: elements 0..7 = K[base..base+7],
  // elements 8..15 = K[base+16..base+23]  (lane>>4 supplies the +8 half)
  v8bf lo = *reinterpret_cast<const v8bf*>(p);
  v8bf hi = *reinterpret_cast<const v8bf*>(p + 16);
  v16bf r;
#pragma unroll
  for (int i = 0; i < 8; ++i) { r[i] = lo[i]; r[i + 8] = hi[i]; }
  return r;
}

static __device__ __forceinline__ v8f wmma_bf16(v16bf a, v16bf b, v8f c) {
  return __builtin_amdgcn_wmma_f32_16x16x32_bf16(false, a, false, b, (short)0, c,
                                                 false, false);
}

// 16-byte global -> LDS copy (async TDM-class path when available)
static __device__ __forceinline__ void cp16_g2s(bf16_t* dst_lds,
                                                const bf16_t* src_g) {
#if HAVE_ASYNC_LDS
  __builtin_amdgcn_global_load_async_to_lds_b128((gptr_v4i)(src_g),
                                                 (lptr_v4i)(dst_lds), 0, 0);
#else
  *reinterpret_cast<uint4*>(dst_lds) = *reinterpret_cast<const uint4*>(src_g);
#endif
}

// ---------------------------------------------------------------------------
// Elementwise / repack kernels
// ---------------------------------------------------------------------------
__global__ void cvt_f32_bf16_kernel(const float* __restrict__ in,
                                    bf16_t* __restrict__ out, size_t n) {
  size_t i = (size_t)blockIdx.x * blockDim.x + threadIdx.x;
  if (i < n) out[i] = (bf16_t)in[i];
}

// in: f32 [K][N] row-major  -> out: bf16 [N][K] (B-transposed for WMMA)
__global__ void transpose_cvt_kernel(const float* __restrict__ in,
                                     bf16_t* __restrict__ out, int K, int N) {
  size_t i = (size_t)blockIdx.x * blockDim.x + threadIdx.x;
  if (i >= (size_t)K * N) return;
  int k = (int)(i / N), n = (int)(i % N);
  out[(size_t)n * K + k] = (bf16_t)in[i];
}

// Wsr f32 [co][ci][kh][kw] -> bf16 [khw][co][ci]   (per-khw Bt = [N=co][K=ci])
__global__ void wsr_repack_kernel(const float* __restrict__ in,
                                  bf16_t* __restrict__ out) {
  size_t i = (size_t)blockIdx.x * blockDim.x + threadIdx.x;  // CC*CC*KHW
  if (i >= (size_t)CC * CC * KHW) return;
  int co = (int)(i >> 13);          // / (CC*KHW) = /8192
  int rem = (int)(i & 8191);
  int ci = rem >> 4;
  int khw = rem & 15;
  out[(size_t)khw * CC * CC + (size_t)co * CC + ci] = (bf16_t)in[i];
}

// kv bf16 [B*NR][2C] -> k bf16 [B][H][NR][HD], vT bf16 [B][H][HD][NR]
__global__ void kv_repack_kernel(const bf16_t* __restrict__ kv,
                                 bf16_t* __restrict__ kout,
                                 bf16_t* __restrict__ vTout) {
  size_t i = (size_t)blockIdx.x * blockDim.x + threadIdx.x;  // (B*NR)*1024
  if (i >= (size_t)BB * NR * 2 * CC) return;
  int row = (int)(i >> 10);
  int j = (int)(i & 1023);
  int b = row >> 8;
  int nr = row & 255;
  bf16_t v = kv[i];
  if (j < CC) {
    int h = j >> 6, d = j & 63;
    kout[((((size_t)b * NHH + h) * NR) + nr) * HD + d] = v;
  } else {
    int j2 = j - CC;
    int h = j2 >> 6, d = j2 & 63;
    vTout[(((size_t)b * NHH + h) * HD + d) * NR + nr] = v;
  }
}

// LayerNorm over 512 channels; f32 in, bf16 out. 1 block(128) per row.
__global__ void layernorm_kernel(const float* __restrict__ XR,
                                 const float* __restrict__ gamma,
                                 const float* __restrict__ beta,
                                 bf16_t* __restrict__ Y) {
  int row = blockIdx.x;                       // B*NR rows
  int tid = threadIdx.x;                      // 128
  const float* x = XR + (size_t)row * CC;
  float4 v = reinterpret_cast<const float4*>(x)[tid];
  float s = v.x + v.y + v.z + v.w;
  float s2 = v.x * v.x + v.y * v.y + v.z * v.z + v.w * v.w;
#pragma unroll
  for (int m = 1; m < 32; m <<= 1) {
    s += __shfl_xor(s, m, 32);
    s2 += __shfl_xor(s2, m, 32);
  }
  __shared__ float sh[2][4];
  int w = tid >> 5;
  if ((tid & 31) == 0) { sh[0][w] = s; sh[1][w] = s2; }
  __syncthreads();
  s = sh[0][0] + sh[0][1] + sh[0][2] + sh[0][3];
  s2 = sh[1][0] + sh[1][1] + sh[1][2] + sh[1][3];
  float mu = s * (1.0f / CC);
  float var = s2 * (1.0f / CC) - mu * mu;
  float rstd = rsqrtf(var + 1e-5f);
  int col = tid * 4;
  float xv[4] = {v.x, v.y, v.z, v.w};
#pragma unroll
  for (int j = 0; j < 4; ++j) {
    float g = gamma[col + j], be = beta[col + j];
    Y[(size_t)row * CC + col + j] = (bf16_t)((xv[j] - mu) * rstd * g + be);
  }
}

// ---------------------------------------------------------------------------
// Generic GEMM: C[M][N] = A[M][K](bf16,row-major) * Bt[N][K](bf16) + bias
// Block = 128 threads (4 waves); block tile 256(M) x 64(N); wave tile 64 x 64
// (4x4 grid of 16x16 WMMA tiles -> 16 WMMAs per 32-deep k-step).
// Shared B tile [64 x 32] is staged to LDS once per k-step with async
// global->LDS copies (double buffered, ASYNCcnt-pipelined).
// ---------------------------------------------------------------------------
template <int OUT_BF16>
__global__ void gemm_bt_kernel(const bf16_t* __restrict__ A,
                               const bf16_t* __restrict__ Bt,
                               const float* __restrict__ bias,
                               void* __restrict__ Cout, int M, int N, int K) {
  __shared__ __align__(16) bf16_t Bs[2][64][32];
  int tid = threadIdx.x;
  int lane = tid & 31;
  int w = tid >> 5;
  int half = lane >> 4;
  int l16 = lane & 15;
  int n0 = blockIdx.x * 64;
  int rowbase = blockIdx.y * 256 + w * 64;

  // stage Bt[n0..n0+63][k0..k0+31] into Bs[buf]; 256 16B chunks / 128 threads
  auto stage = [&](int buf, int k0) {
#pragma unroll
    for (int j = 0; j < 2; ++j) {
      int c = tid * 2 + j;        // 0..255
      int row = c >> 2;           // 0..63
      int kc = c & 3;             // 0..3, 8 bf16 each
      cp16_g2s(&Bs[buf][row][kc * 8],
               Bt + (size_t)(n0 + row) * K + k0 + kc * 8);
    }
  };

  v8f acc[4][4] = {};
  stage(0, 0);
#pragma unroll 1
  for (int k0 = 0; k0 < K; k0 += 32) {
    int cur = (k0 >> 5) & 1;
    bool more = (k0 + 32) < K;
    if (more) stage(cur ^ 1, k0 + 32);
#if HAVE_ASYNC_LDS
    if (more)
      asm volatile("s_wait_asynccnt 0x2" ::: "memory");
    else
      asm volatile("s_wait_asynccnt 0x0" ::: "memory");
#endif
    __syncthreads();

    v16bf bfr[4];
#pragma unroll
    for (int t = 0; t < 4; ++t)
      bfr[t] = load_frag(&Bs[cur][t * 16 + l16][half * 8]);
#pragma unroll
    for (int im = 0; im < 4; ++im) {
      v16bf af =
          load_frag(A + (size_t)(rowbase + im * 16 + l16) * K + k0 + half * 8);
#pragma unroll
      for (int t = 0; t < 4; ++t)
        acc[im][t] = wmma_bf16(af, bfr[t], acc[im][t]);
    }
    __syncthreads();
  }

#pragma unroll
  for (int im = 0; im < 4; ++im)
#pragma unroll
    for (int t = 0; t < 4; ++t) {
      int col = n0 + t * 16 + l16;
      float bv = bias[col];
#pragma unroll
      for (int r = 0; r < 8; ++r) {
        int row = rowbase + im * 16 + half * 8 + r;
        float v = acc[im][t][r] + bv;
        if (OUT_BF16)
          reinterpret_cast<bf16_t*>(Cout)[(size_t)row * N + col] = (bf16_t)v;
        else
          reinterpret_cast<float*>(Cout)[(size_t)row * N + col] = v;
      }
    }
}

// ---------------------------------------------------------------------------
// Spatial-reduction conv as accumulated GEMM with im2col row remap.
// grid (N/64=8, M/64=4, B); xr[b][p][co] = sum_khw sum_ci x[b][n(p,khw)][ci]
//                                          * WsrT[khw][co][ci]  + bsr[co]
// Smaller 16x64 wave tile: keeps 256 blocks in flight (M is only 256/batch).
// ---------------------------------------------------------------------------
__global__ void sr_gemm_kernel(const bf16_t* __restrict__ Xbf,
                               const bf16_t* __restrict__ WsrT,
                               const float* __restrict__ bsr,
                               float* __restrict__ XR) {
  int lane = threadIdx.x & 31;
  int w = threadIdx.x >> 5;
  int half = lane >> 4;
  int l16 = lane & 15;
  int b = blockIdx.z;
  int n0 = blockIdx.x * 64;
  int pbase = blockIdx.y * 64 + w * 16;
  int p = pbase + l16;
  int ph = p >> 4, pw = p & 15;

  const bf16_t* x = Xbf + (size_t)b * NN * CC;
  v8f acc[4] = {};
#pragma unroll 1
  for (int khw = 0; khw < KHW; ++khw) {
    int kh = khw >> 2, kw = khw & 3;
    int nrow = (4 * ph + kh) * 64 + 4 * pw + kw;
    const bf16_t* aptr = x + (size_t)nrow * CC + half * 8;
    const bf16_t* wb = WsrT + (size_t)khw * CC * CC;
    for (int k0 = 0; k0 < CC; k0 += 32) {
      v16bf af = load_frag(aptr + k0);
#pragma unroll
      for (int t = 0; t < 4; ++t) {
        v16bf bfg =
            load_frag(wb + (size_t)(n0 + t * 16 + l16) * CC + k0 + half * 8);
        acc[t] = wmma_bf16(af, bfg, acc[t]);
      }
    }
  }
#pragma unroll
  for (int t = 0; t < 4; ++t) {
    int col = n0 + t * 16 + l16;
    float bv = bsr[col];
#pragma unroll
    for (int r = 0; r < 8; ++r) {
      int prow = pbase + half * 8 + r;
      XR[((size_t)b * NR + prow) * CC + col] = acc[t][r] + bv;
    }
  }
}

// ---------------------------------------------------------------------------
// Attention: per (b,h): S = Q[4096x64] Kt -> softmax -> * V.
// Block 128 threads (4 waves); wave owns 16 query rows, full Nr=256 scores.
// K stored [NR][HD] is natively the Bt layout for Q.K^T; V stored transposed
// [HD][NR] for P.V. P round-trips LDS to convert C-layout -> A-fragment.
// ---------------------------------------------------------------------------
__global__ void attn_kernel(const bf16_t* __restrict__ Q,
                            const bf16_t* __restrict__ Km,
                            const bf16_t* __restrict__ Vt,
                            bf16_t* __restrict__ O) {
  __shared__ __align__(16) bf16_t Plds[4][16][NR + 8];
  int lane = threadIdx.x & 31;
  int w = threadIdx.x >> 5;
  int half = lane >> 4;
  int l16 = lane & 15;
  int b = blockIdx.z, h = blockIdx.y;
  int qbase = blockIdx.x * 64 + w * 16;

  const bf16_t* qp =
      Q + (size_t)(b * NN + qbase + l16) * CC + h * HD + half * 8;
  const bf16_t* kp = Km + ((size_t)b * NHH + h) * NR * HD;
  const bf16_t* vp = Vt + ((size_t)b * NHH + h) * HD * NR;

  // S = Q K^T : 16 n-tiles x 2 k-steps
  v8f s[16] = {};
#pragma unroll
  for (int ks = 0; ks < 2; ++ks) {
    v16bf af = load_frag(qp + ks * 32);
#pragma unroll
    for (int t = 0; t < 16; ++t) {
      v16bf bfg =
          load_frag(kp + (size_t)(t * 16 + l16) * HD + ks * 32 + half * 8);
      s[t] = wmma_bf16(af, bfg, s[t]);
    }
  }

  // softmax over 256 per row (rows split across half-waves in C layout)
  const float scale = 0.125f;  // hd^-0.5
  float mx[8], sm[8];
#pragma unroll
  for (int r = 0; r < 8; ++r) {
    mx[r] = -3.4e38f;
#pragma unroll
    for (int t = 0; t < 16; ++t) mx[r] = fmaxf(mx[r], s[t][r]);
  }
#pragma unroll
  for (int m = 1; m < 16; m <<= 1)
#pragma unroll
    for (int r = 0; r < 8; ++r) mx[r] = fmaxf(mx[r], __shfl_xor(mx[r], m, 32));
#pragma unroll
  for (int r = 0; r < 8; ++r) sm[r] = 0.f;
#pragma unroll
  for (int t = 0; t < 16; ++t)
#pragma unroll
    for (int r = 0; r < 8; ++r) {
      float e = __expf((s[t][r] - mx[r]) * scale);
      s[t][r] = e;
      sm[r] += e;
    }
#pragma unroll
  for (int m = 1; m < 16; m <<= 1)
#pragma unroll
    for (int r = 0; r < 8; ++r) sm[r] += __shfl_xor(sm[r], m, 32);
#pragma unroll
  for (int r = 0; r < 8; ++r) sm[r] = 1.0f / sm[r];

  // store P to LDS (C-layout scatter), re-read in A-fragment layout
#pragma unroll
  for (int t = 0; t < 16; ++t)
#pragma unroll
    for (int r = 0; r < 8; ++r)
      Plds[w][half * 8 + r][t * 16 + l16] = (bf16_t)(s[t][r] * sm[r]);
  asm volatile("s_wait_dscnt 0" ::: "memory");

  // O = P V : 4 n-tiles x 8 k-steps
  v8f o[4] = {};
#pragma unroll
  for (int ks = 0; ks < 8; ++ks) {
    v16bf af = load_frag(&Plds[w][l16][ks * 32 + half * 8]);
#pragma unroll
    for (int tn = 0; tn < 4; ++tn) {
      v16bf bfg =
          load_frag(vp + (size_t)(tn * 16 + l16) * NR + ks * 32 + half * 8);
      o[tn] = wmma_bf16(af, bfg, o[tn]);
    }
  }
#pragma unroll
  for (int tn = 0; tn < 4; ++tn) {
    int col = h * HD + tn * 16 + l16;
#pragma unroll
    for (int r = 0; r < 8; ++r) {
      int row = b * NN + qbase + half * 8 + r;
      O[(size_t)row * CC + col] = (bf16_t)o[tn][r];
    }
  }
}

// ---------------------------------------------------------------------------
// Host-side launch
// ---------------------------------------------------------------------------
extern "C" void kernel_launch(void* const* d_in, const int* in_sizes, int n_in,
                              void* d_out, int out_size, void* d_ws,
                              size_t ws_size, hipStream_t stream) {
  (void)in_sizes; (void)n_in; (void)out_size; (void)ws_size;
  const float* x = (const float*)d_in[0];
  const float* Wq = (const float*)d_in[1];
  const float* bq = (const float*)d_in[2];
  const float* Wkv = (const float*)d_in[3];
  const float* bkv = (const float*)d_in[4];
  const float* Wp = (const float*)d_in[5];
  const float* bp = (const float*)d_in[6];
  const float* Wsr = (const float*)d_in[7];
  const float* bsr = (const float*)d_in[8];
  const float* gamma = (const float*)d_in[9];
  const float* beta = (const float*)d_in[10];
  float* out = (float*)d_out;

  char* ws = (char*)d_ws;
  size_t off = 0;
  auto take = [&](size_t bytes) {
    void* p = ws + off;
    off = (off + bytes + 255) & ~(size_t)255;
    return p;
  };
  bf16_t* x_bf   = (bf16_t*)take((size_t)BB * NN * CC * 2);       // 32 MB
  bf16_t* q_bf   = (bf16_t*)take((size_t)BB * NN * CC * 2);       // 32 MB
  bf16_t* ao_bf  = (bf16_t*)take((size_t)BB * NN * CC * 2);       // 32 MB
  bf16_t* WqT    = (bf16_t*)take((size_t)CC * CC * 2);
  bf16_t* WkvT   = (bf16_t*)take((size_t)2 * CC * CC * 2);
  bf16_t* WpT    = (bf16_t*)take((size_t)CC * CC * 2);
  bf16_t* WsrT   = (bf16_t*)take((size_t)KHW * CC * CC * 2);      // 8 MB
  float*  xr     = (float*) take((size_t)BB * NR * CC * 4);
  bf16_t* xrln   = (bf16_t*)take((size_t)BB * NR * CC * 2);
  bf16_t* kv     = (bf16_t*)take((size_t)BB * NR * 2 * CC * 2);
  bf16_t* k_bf   = (bf16_t*)take((size_t)BB * NHH * NR * HD * 2);
  bf16_t* vT_bf  = (bf16_t*)take((size_t)BB * NHH * HD * NR * 2);

  // 1) precision conversions / weight repacks
  {
    size_t n = (size_t)BB * NN * CC;
    cvt_f32_bf16_kernel<<<dim3((unsigned)((n + 255) / 256)), 256, 0, stream>>>(
        x, x_bf, n);
  }
  transpose_cvt_kernel<<<dim3((CC * CC + 255) / 256), 256, 0, stream>>>(
      Wq, WqT, CC, CC);
  transpose_cvt_kernel<<<dim3((CC * 2 * CC + 255) / 256), 256, 0, stream>>>(
      Wkv, WkvT, CC, 2 * CC);
  transpose_cvt_kernel<<<dim3((CC * CC + 255) / 256), 256, 0, stream>>>(
      Wp, WpT, CC, CC);
  wsr_repack_kernel<<<dim3((CC * CC * KHW + 255) / 256), 256, 0, stream>>>(
      Wsr, WsrT);

  // 2) Q projection: [B*N,512] x [512,512] -> q_bf
  gemm_bt_kernel<1><<<dim3(CC / 64, (BB * NN) / 256), 128, 0, stream>>>(
      x_bf, WqT, bq, q_bf, BB * NN, CC, CC);

  // 3) spatial reduction conv -> xr (f32)
  sr_gemm_kernel<<<dim3(CC / 64, NR / 64, BB), 128, 0, stream>>>(
      x_bf, WsrT, bsr, xr);

  // 4) LayerNorm -> xrln (bf16)
  layernorm_kernel<<<dim3(BB * NR), 128, 0, stream>>>(xr, gamma, beta, xrln);

  // 5) KV projection: [B*NR,512] x [512,1024] -> kv
  gemm_bt_kernel<1><<<dim3((2 * CC) / 64, (BB * NR) / 256), 128, 0, stream>>>(
      xrln, WkvT, bkv, kv, BB * NR, 2 * CC, CC);

  // 6) split kv -> k [B,H,NR,HD], vT [B,H,HD,NR]
  {
    size_t n = (size_t)BB * NR * 2 * CC;
    kv_repack_kernel<<<dim3((unsigned)((n + 255) / 256)), 256, 0, stream>>>(
        kv, k_bf, vT_bf);
  }

  // 7) attention -> ao_bf
  attn_kernel<<<dim3(NN / 64, NHH, BB), 128, 0, stream>>>(q_bf, k_bf, vT_bf,
                                                          ao_bf);

  // 8) output projection: [B*N,512] x [512,512] -> out (f32)
  gemm_bt_kernel<0><<<dim3(CC / 64, (BB * NN) / 256), 128, 0, stream>>>(
      ao_bf, WpT, bp, out, BB * NN, CC, CC);
}
